// MoEFFN_23021024706754
// MI455X (gfx1250) — matrix operand order repaired and checked by reading, not compiled, not run on hardware
//
#include <hip/hip_runtime.h>
#include <hip/hip_bf16.h>
#include <math.h>

// MoE top-1 FFN for MI455X (gfx1250, wave32, WMMA bf16 16x16x32).
// B=4 L=1024 D=1024 H=4096 E=8 -> N=4096 tokens, ~69 GFLOP, 256MB fp32 weights.
// Strategy: gate -> per-expert token lists -> two bf16 WMMA GEMM passes with
// f32->bf16 conversion in the LDS staging (pre-packed into the documented
// CDNA5 A/B fragment layouts so fragment loads are ds_load_b128 pairs).

#define NTOK 4096
#define DDIM 1024
#define HDIM 4096
#define NEXP 8
#define AUXC 0.01f

typedef __attribute__((ext_vector_type(16))) __bf16 v16bf;
typedef __attribute__((ext_vector_type(8)))  float  v8f;

union FragBF { int4 q[2]; v16bf v; };

// round-to-nearest-even f32 -> bf16, packed pair (lo = a, hi = b)
__device__ __forceinline__ unsigned pk2bf(float a, float b) {
  unsigned ua = __builtin_bit_cast(unsigned, a);
  unsigned ub = __builtin_bit_cast(unsigned, b);
  ua += 0x7fffu + ((ua >> 16) & 1u);
  ub += 0x7fffu + ((ub >> 16) & 1u);
  return (ua >> 16) | (ub & 0xffff0000u);
}
__device__ __forceinline__ unsigned short f2bf(float a) {
  unsigned ua = __builtin_bit_cast(unsigned, a);
  ua += 0x7fffu + ((ua >> 16) & 1u);
  return (unsigned short)(ua >> 16);
}

// ---------------- kernel 0: zero expert counters -------------------------
__global__ void moe_zero_kernel(int* cnt) {
  if (threadIdx.x < NEXP) cnt[threadIdx.x] = 0;
}

// ---------------- kernel 1: gating (one wave per token) ------------------
__global__ __launch_bounds__(256) void moe_gate_kernel(
    const float* __restrict__ x, const float* __restrict__ Wg,
    const float* __restrict__ bg, float* __restrict__ gprobs,
    int* __restrict__ cnt, int* __restrict__ perm) {
  const int w = threadIdx.x >> 5;
  const int lane = threadIdx.x & 31;
  const int n = blockIdx.x * 8 + w;  // 512 blocks * 8 waves = 4096 tokens
  float acc[NEXP];
#pragma unroll
  for (int e = 0; e < NEXP; ++e) acc[e] = 0.f;
  const float* xr = x + (size_t)n * DDIM;
  for (int d = lane; d < DDIM; d += 32) {
    float xv = xr[d];
    const float4* wr = (const float4*)(Wg + (size_t)d * NEXP);
    float4 w0 = wr[0], w1 = wr[1];
    acc[0] += xv * w0.x; acc[1] += xv * w0.y;
    acc[2] += xv * w0.z; acc[3] += xv * w0.w;
    acc[4] += xv * w1.x; acc[5] += xv * w1.y;
    acc[6] += xv * w1.z; acc[7] += xv * w1.w;
  }
#pragma unroll
  for (int off = 16; off; off >>= 1)
#pragma unroll
    for (int e = 0; e < NEXP; ++e) acc[e] += __shfl_xor(acc[e], off, 32);
  if (lane == 0) {
    float lg[NEXP]; float mx = -1e30f;
#pragma unroll
    for (int e = 0; e < NEXP; ++e) { lg[e] = acc[e] + bg[e]; mx = fmaxf(mx, lg[e]); }
    // argmax (first max, same tie rule as jnp.argmax) on logits
    int best = 0; float bv = lg[0];
#pragma unroll
    for (int e = 1; e < NEXP; ++e) if (lg[e] > bv) { bv = lg[e]; best = e; }
    float s = 0.f;
#pragma unroll
    for (int e = 0; e < NEXP; ++e) { lg[e] = __expf(lg[e] - mx); s += lg[e]; }
    float inv = 1.f / s;
#pragma unroll
    for (int e = 0; e < NEXP; ++e) gprobs[(size_t)n * NEXP + e] = lg[e] * inv;
    int pos = atomicAdd(&cnt[best], 1);
    perm[best * NTOK + pos] = n;
  }
}

// ---------------- kernel 2: deterministic P/C/aux reduction --------------
__global__ __launch_bounds__(256) void moe_stats_kernel(
    const float* __restrict__ gprobs, const int* __restrict__ cnt,
    float* __restrict__ tail /* d_out + N*D : [aux, P(8), C(8)] */) {
  __shared__ float red[256][NEXP + 1];
  const int t = threadIdx.x;
  float p[NEXP];
#pragma unroll
  for (int e = 0; e < NEXP; ++e) p[e] = 0.f;
  for (int n = t; n < NTOK; n += 256)
#pragma unroll
    for (int e = 0; e < NEXP; ++e) p[e] += gprobs[(size_t)n * NEXP + e];
#pragma unroll
  for (int e = 0; e < NEXP; ++e) red[t][e] = p[e];
  __syncthreads();
  for (int s = 128; s; s >>= 1) {
    if (t < s)
#pragma unroll
      for (int e = 0; e < NEXP; ++e) red[t][e] += red[t + s][e];
    __syncthreads();
  }
  if (t < NEXP) {
    tail[1 + t] = red[0][t] / (float)NTOK;
    tail[1 + NEXP + t] = (float)cnt[t] / (float)NTOK;
  }
  if (t == 0) {
    float aux = 0.f;
#pragma unroll
    for (int e = 0; e < NEXP; ++e)
      aux += (red[0][e] / (float)NTOK) * ((float)cnt[e] / (float)NTOK);
    tail[0] = (float)NEXP * aux * AUXC;
  }
}

// ---------------- GEMM pass 1: h = gelu(gather(x) @ W1[e] + b1[e]) -------
// block tile 128(M) x 128(N), 8 waves, each wave 32x64 (2x4 16x16 tiles), K=32.
__global__ __launch_bounds__(256) void moe_ffn1_kernel(
    const float* __restrict__ x, const float* __restrict__ W1,
    const float* __restrict__ b1, const int* __restrict__ cnt,
    const int* __restrict__ perm, unsigned short* __restrict__ hbuf) {
  __shared__ int tokLds[128];
  __shared__ unsigned aLds[128 * 16];  // [m][kpair]  bf16 pairs
  __shared__ unsigned bLds[128 * 16];  // [n][kpair]  bf16 pairs

  const int e = blockIdx.z;
  const int Ne = cnt[e];
  const int mbase = blockIdx.x * 128;
  if (mbase >= Ne) return;
  const int nbase = blockIdx.y * 128;
  const int t = threadIdx.x;
  const int lane = t & 31;
  const int w = t >> 5;
  const int wm = w & 3;   // 4 wave-rows of 32
  const int wn = w >> 2;  // 2 wave-cols of 64

  if (t < 128) {
    int pos = mbase + t;
    tokLds[t] = (pos < Ne) ? perm[e * NTOK + pos] : -1;
  }
  __syncthreads();

  v8f acc[2][4] = {};

  const int sm_t = t >> 1, half_t = t & 1;   // A staging role
  const int kp_t = t >> 4, n0_t = (t & 15) * 8;  // B staging role
  int tokA = tokLds[sm_t]; if (tokA < 0) tokA = 0;
  const size_t wbase = (size_t)e * DDIM * HDIM;

  for (int k0 = 0; k0 < DDIM; k0 += 32) {
    // --- stage A: gathered x rows, f32 -> bf16 pair-packed ---
    {
      const float4* src = (const float4*)(x + (size_t)tokA * DDIM + k0 + half_t * 16);
      float4 f0 = src[0], f1 = src[1], f2 = src[2], f3 = src[3];
      int4 p0 = make_int4((int)pk2bf(f0.x, f0.y), (int)pk2bf(f0.z, f0.w),
                          (int)pk2bf(f1.x, f1.y), (int)pk2bf(f1.z, f1.w));
      int4 p1 = make_int4((int)pk2bf(f2.x, f2.y), (int)pk2bf(f2.z, f2.w),
                          (int)pk2bf(f3.x, f3.y), (int)pk2bf(f3.z, f3.w));
      int4* dst = (int4*)(aLds + sm_t * 16 + half_t * 8);
      dst[0] = p0; dst[1] = p1;
    }
    // --- stage B: W1[e][k0+2kp .. +1][nbase+n0 ..+8], pack k-pairs ---
    {
      const float* r0 = W1 + wbase + (size_t)(k0 + 2 * kp_t) * HDIM + nbase + n0_t;
      const float* r1 = r0 + HDIM;
      float4 a0 = *(const float4*)r0, a1 = *(const float4*)(r0 + 4);
      float4 c0 = *(const float4*)r1, c1 = *(const float4*)(r1 + 4);
      unsigned* bl = bLds + n0_t * 16 + kp_t;
      bl[0 * 16] = pk2bf(a0.x, c0.x); bl[1 * 16] = pk2bf(a0.y, c0.y);
      bl[2 * 16] = pk2bf(a0.z, c0.z); bl[3 * 16] = pk2bf(a0.w, c0.w);
      bl[4 * 16] = pk2bf(a1.x, c1.x); bl[5 * 16] = pk2bf(a1.y, c1.y);
      bl[6 * 16] = pk2bf(a1.z, c1.z); bl[7 * 16] = pk2bf(a1.w, c1.w);
    }
    __syncthreads();

    // --- fragments per documented CDNA5 bf16 layouts + WMMA ---
    const int khA = (lane >> 4) * 4;  // A: dword offset per K-half
    const int khB = (lane >> 4) * 4;  // B: dword offset per K-half
    FragBF fa[2];
#pragma unroll
    for (int sm = 0; sm < 2; ++sm) {
      const unsigned* ar = aLds + (wm * 32 + sm * 16 + (lane & 15)) * 16;
      fa[sm].q[0] = *(const int4*)(ar + khA);
      fa[sm].q[1] = *(const int4*)(ar + 8 + khA);
    }
    FragBF fb[4];
#pragma unroll
    for (int sn = 0; sn < 4; ++sn) {
      const unsigned* br = bLds + (wn * 64 + sn * 16 + (lane & 15)) * 16;
      fb[sn].q[0] = *(const int4*)(br + khB);
      fb[sn].q[1] = *(const int4*)(br + 8 + khB);
    }
#pragma unroll
    for (int sm = 0; sm < 2; ++sm)
#pragma unroll
      for (int sn = 0; sn < 4; ++sn)
        acc[sm][sn] = __builtin_amdgcn_wmma_f32_16x16x32_bf16(
            false, fa[sm].v, false, fb[sn].v, (short)0, acc[sm][sn],
            false, false);
    __syncthreads();
  }

  // --- epilogue: +bias, exact GELU, scatter bf16 rows of hbuf ---
  const int colLane = lane & 15;
  const int rowHalf = (lane >> 4) * 8;
#pragma unroll
  for (int sm = 0; sm < 2; ++sm)
#pragma unroll
    for (int sn = 0; sn < 4; ++sn) {
      const int gn = nbase + wn * 64 + sn * 16 + colLane;
      const float bias = b1[(size_t)e * HDIM + gn];
#pragma unroll
      for (int r = 0; r < 8; ++r) {
        const int ml = wm * 32 + sm * 16 + rowHalf + r;
        const int tok = tokLds[ml];
        if (tok >= 0) {
          float v = acc[sm][sn][r] + bias;
          v = 0.5f * v * (1.0f + erff(v * 0.70710678118654752f));
          hbuf[(size_t)tok * HDIM + gn] = f2bf(v);
        }
      }
    }
}

// ---------------- GEMM pass 2: out = gather(h) @ W2[e] + b2[e] -----------
__global__ __launch_bounds__(256) void moe_ffn2_kernel(
    const unsigned short* __restrict__ hbuf, const float* __restrict__ W2,
    const float* __restrict__ b2, const int* __restrict__ cnt,
    const int* __restrict__ perm, float* __restrict__ out) {
  __shared__ int tokLds[128];
  __shared__ unsigned aLds[128 * 16];
  __shared__ unsigned bLds[128 * 16];

  const int e = blockIdx.z;
  const int Ne = cnt[e];
  const int mbase = blockIdx.x * 128;
  if (mbase >= Ne) return;
  const int nbase = blockIdx.y * 128;
  const int t = threadIdx.x;
  const int lane = t & 31;
  const int w = t >> 5;
  const int wm = w & 3;
  const int wn = w >> 2;

  if (t < 128) {
    int pos = mbase + t;
    tokLds[t] = (pos < Ne) ? perm[e * NTOK + pos] : -1;
  }
  __syncthreads();

  v8f acc[2][4] = {};

  const int sm_t = t >> 1, half_t = t & 1;
  const int kp_t = t >> 4, n0_t = (t & 15) * 8;
  int tokA = tokLds[sm_t]; if (tokA < 0) tokA = 0;
  const size_t wbase = (size_t)e * HDIM * DDIM;

  for (int k0 = 0; k0 < HDIM; k0 += 32) {
    // --- stage A: hbuf already bf16, row-major == pair-packed ---
    {
      const int4* src =
          (const int4*)(hbuf + (size_t)tokA * HDIM + k0 + half_t * 16);
      int4* dst = (int4*)(aLds + sm_t * 16 + half_t * 8);
      dst[0] = src[0]; dst[1] = src[1];
    }
    // --- stage B: W2[e] f32 -> bf16 pair-packed ---
    {
      const float* r0 = W2 + wbase + (size_t)(k0 + 2 * kp_t) * DDIM + nbase + n0_t;
      const float* r1 = r0 + DDIM;
      float4 a0 = *(const float4*)r0, a1 = *(const float4*)(r0 + 4);
      float4 c0 = *(const float4*)r1, c1 = *(const float4*)(r1 + 4);
      unsigned* bl = bLds + n0_t * 16 + kp_t;
      bl[0 * 16] = pk2bf(a0.x, c0.x); bl[1 * 16] = pk2bf(a0.y, c0.y);
      bl[2 * 16] = pk2bf(a0.z, c0.z); bl[3 * 16] = pk2bf(a0.w, c0.w);
      bl[4 * 16] = pk2bf(a1.x, c1.x); bl[5 * 16] = pk2bf(a1.y, c1.y);
      bl[6 * 16] = pk2bf(a1.z, c1.z); bl[7 * 16] = pk2bf(a1.w, c1.w);
    }
    __syncthreads();

    const int khA = (lane >> 4) * 4;
    const int khB = (lane >> 4) * 4;
    FragBF fa[2];
#pragma unroll
    for (int sm = 0; sm < 2; ++sm) {
      const unsigned* ar = aLds + (wm * 32 + sm * 16 + (lane & 15)) * 16;
      fa[sm].q[0] = *(const int4*)(ar + khA);
      fa[sm].q[1] = *(const int4*)(ar + 8 + khA);
    }
    FragBF fb[4];
#pragma unroll
    for (int sn = 0; sn < 4; ++sn) {
      const unsigned* br = bLds + (wn * 64 + sn * 16 + (lane & 15)) * 16;
      fb[sn].q[0] = *(const int4*)(br + khB);
      fb[sn].q[1] = *(const int4*)(br + 8 + khB);
    }
#pragma unroll
    for (int sm = 0; sm < 2; ++sm)
#pragma unroll
      for (int sn = 0; sn < 4; ++sn)
        acc[sm][sn] = __builtin_amdgcn_wmma_f32_16x16x32_bf16(
            false, fa[sm].v, false, fb[sn].v, (short)0, acc[sm][sn],
            false, false);
    __syncthreads();
  }

  const int colLane = lane & 15;
  const int rowHalf = (lane >> 4) * 8;
#pragma unroll
  for (int sm = 0; sm < 2; ++sm)
#pragma unroll
    for (int sn = 0; sn < 4; ++sn) {
      const int gn = nbase + wn * 64 + sn * 16 + colLane;
      const float bias = b2[(size_t)e * DDIM + gn];
#pragma unroll
      for (int r = 0; r < 8; ++r) {
        const int ml = wm * 32 + sm * 16 + rowHalf + r;
        const int tok = tokLds[ml];
        if (tok >= 0) out[(size_t)tok * DDIM + gn] = acc[sm][sn][r] + bias;
      }
    }
}

// ---------------- host launcher ------------------------------------------
extern "C" void kernel_launch(void* const* d_in, const int* in_sizes, int n_in,
                              void* d_out, int out_size, void* d_ws,
                              size_t ws_size, hipStream_t stream) {
  const float* x  = (const float*)d_in[0];
  const float* Wg = (const float*)d_in[1];
  const float* bg = (const float*)d_in[2];
  const float* W1 = (const float*)d_in[3];
  const float* b1 = (const float*)d_in[4];
  const float* W2 = (const float*)d_in[5];
  const float* b2 = (const float*)d_in[6];
  float* out = (float*)d_out;

  // workspace layout (~33.8 MB total)
  char* ws = (char*)d_ws;
  int* cnt = (int*)ws;                                   // 32 B
  int* perm = (int*)(ws + 32);                           // E*N*4 = 128 KB
  float* gprobs = (float*)(ws + 32 + NEXP * NTOK * 4);   // N*E*4 = 128 KB
  unsigned short* hbuf = (unsigned short*)(ws + 262400); // N*H*2 = 32 MB

  moe_zero_kernel<<<1, 32, 0, stream>>>(cnt);
  moe_gate_kernel<<<NTOK / 8, 256, 0, stream>>>(x, Wg, bg, gprobs, cnt, perm);
  moe_stats_kernel<<<1, 256, 0, stream>>>(gprobs, cnt,
                                          out + (size_t)NTOK * DDIM);
  moe_ffn1_kernel<<<dim3(NTOK / 128, HDIM / 128, NEXP), 256, 0, stream>>>(
      x, W1, b1, cnt, perm, hbuf);
  moe_ffn2_kernel<<<dim3(NTOK / 128, DDIM / 128, NEXP), 256, 0, stream>>>(
      hbuf, W2, b2, cnt, perm, out);
}
